// tTanhModel_996432413177
// MI455X (gfx1250) — matrix operand order repaired and checked by reading, not compiled, not run on hardware
//
#include <hip/hip_runtime.h>
#include <stdint.h>

#define B_ 64
#define S_ 512
#define I_ 256
#define H_ 1024
#define O_ 256

typedef __attribute__((ext_vector_type(16))) __bf16 v16bf;
typedef __attribute__((ext_vector_type(8)))  float  v8f;
typedef __attribute__((ext_vector_type(8)))  unsigned int v8u;
typedef __attribute__((ext_vector_type(4)))  unsigned int u32x4;
typedef __attribute__((ext_vector_type(8)))  int i32x8;

// ---------------- helpers ----------------

__device__ __forceinline__ unsigned short f32_to_bf16(float f) {
  unsigned int u = __builtin_bit_cast(unsigned int, f);
  u += 0x7FFFu + ((u >> 16) & 1u);   // round-to-nearest-even
  return (unsigned short)(u >> 16);
}

// A fragment (16x32 bf16), A row-major with leading dim lda.
// ISA layout: lanes 0-15 -> rows M=0..15, K = {0..7}+8*half and {16..23}+8*half
__device__ __forceinline__ v16bf load_frag_a(const unsigned short* A, int lda,
                                             int mbase, int kbase, int lane) {
  const int half = lane >> 4;
  const int m = mbase + (lane & 15);
  const unsigned short* row = A + (size_t)m * lda + kbase + 8 * half;
  uint4 lo = *(const uint4*)(row);        // K = 0..7  (+8*half)
  uint4 hi = *(const uint4*)(row + 16);   // K = 16..23 (+8*half)
  v8u u = { lo.x, lo.y, lo.z, lo.w, hi.x, hi.y, hi.z, hi.w };
  return __builtin_bit_cast(v16bf, u);
}

// B fragment (32x16 bf16) where B(k,n) = W[nbase+n][kbase+k], W row-major ldw.
// ISA layout: lane holds N=lane&15, K = {0..7}+8*half and {16..23}+8*half
__device__ __forceinline__ v16bf load_frag_b(const unsigned short* W, int ldw,
                                             int nbase, int kbase, int lane) {
  const int half = lane >> 4;
  const int n = nbase + (lane & 15);
  const unsigned short* row = W + (size_t)n * ldw + kbase + 8 * half;
  uint4 lo = *(const uint4*)(row);        // K = 0..7  (+8*half)
  uint4 hi = *(const uint4*)(row + 16);   // K = 16..23 (+8*half)
  v8u u = { lo.x, lo.y, lo.z, lo.w, hi.x, hi.y, hi.z, hi.w };
  return __builtin_bit_cast(v16bf, u);
}

// Async global->LDS copy of 16 bytes (tracked by ASYNCcnt, bypasses VGPRs).
__device__ __forceinline__ void async_b128(unsigned lds_off, const void* g) {
  unsigned long long ga = (unsigned long long)(uintptr_t)g;
  asm volatile("global_load_async_to_lds_b128 %0, %1, off"
               :: "v"(lds_off), "v"(ga) : "memory");
}
__device__ __forceinline__ void wait_async0() {
  asm volatile("s_wait_asynccnt 0x0" ::: "memory");
}

__device__ __forceinline__ void grid_sync(unsigned int* cnt, unsigned int target) {
  __syncthreads();
  __builtin_amdgcn_s_cluster_barrier();   // NOP unless dispatched as a cluster
  if (threadIdx.x == 0) {
    atomicAdd(cnt, 1u);
    while (__hip_atomic_load(cnt, __ATOMIC_ACQUIRE, __HIP_MEMORY_SCOPE_AGENT) < target) {
      __builtin_amdgcn_s_sleep(1);
    }
  }
  __syncthreads();
}

// ---------------- kernels ----------------

__global__ void cvt_f32_bf16_kernel(const float* __restrict__ src,
                                    unsigned short* __restrict__ dst, size_t n) {
  size_t i = (size_t)blockIdx.x * blockDim.x + threadIdx.x;
  size_t stride = (size_t)gridDim.x * blockDim.x;
  for (; i < n; i += stride) dst[i] = f32_to_bf16(src[i]);
}

// C[M,N] = A[M,K](bf16) * W[N,K]^T(bf16) + bias1[N] + bias2[N], C f32.
// Block tile 128x64, 8 waves, each wave owns a 2x2 grid of 16x16 accumulators.
// K staged 32 at a time into double-buffered LDS via async global->LDS copies.
__global__ __launch_bounds__(256) void gemm_bf16_kernel(
    const unsigned short* __restrict__ A, int lda,
    const unsigned short* __restrict__ W, int ldb,
    float* __restrict__ C, int ldc, int K,
    const float* __restrict__ bias1, const float* __restrict__ bias2) {
  __shared__ unsigned short Als[2][128 * 32];
  __shared__ unsigned short Bls[2][64 * 32];

  const int tid  = threadIdx.x;
  const int wave = tid >> 5;
  const int lane = tid & 31;
  const size_t mwg = (size_t)blockIdx.y * 128;
  const size_t nwg = (size_t)blockIdx.x * 64;

  const int wa = wave >> 1;   // 0..3 -> m-tile pair
  const int wb = wave & 1;    // 0..1 -> n-tile pair
  v8f acc[2][2] = {};

  auto stage = [&](int kb, int buf) {
    unsigned aL = (unsigned)(uintptr_t)&Als[buf][0];
    unsigned bL = (unsigned)(uintptr_t)&Bls[buf][0];
    const int i0 = tid, i1 = tid + 256;
    async_b128(aL + i0 * 16, A + (mwg + (i0 >> 2)) * lda + kb + (i0 & 3) * 8);
    async_b128(aL + i1 * 16, A + (mwg + (i1 >> 2)) * lda + kb + (i1 & 3) * 8);
    async_b128(bL + i0 * 16, W + (nwg + (i0 >> 2)) * ldb + kb + (i0 & 3) * 8);
  };

  stage(0, 0);
  wait_async0();
  __syncthreads();

  const int nchunks = K >> 5;
  for (int c = 0; c < nchunks; ++c) {
    const int cur = c & 1;
    if (c + 1 < nchunks) stage((c + 1) << 5, cur ^ 1);  // overlap with compute

    v16bf a0 = load_frag_a(&Als[cur][0], 32, (2 * wa + 0) * 16, 0, lane);
    v16bf a1 = load_frag_a(&Als[cur][0], 32, (2 * wa + 1) * 16, 0, lane);
    v16bf b0 = load_frag_b(&Bls[cur][0], 32, (2 * wb + 0) * 16, 0, lane);
    v16bf b1 = load_frag_b(&Bls[cur][0], 32, (2 * wb + 1) * 16, 0, lane);
    acc[0][0] = __builtin_amdgcn_wmma_f32_16x16x32_bf16(false, a0, false, b0, (short)0, acc[0][0], false, false);
    acc[0][1] = __builtin_amdgcn_wmma_f32_16x16x32_bf16(false, a0, false, b1, (short)0, acc[0][1], false, false);
    acc[1][0] = __builtin_amdgcn_wmma_f32_16x16x32_bf16(false, a1, false, b0, (short)0, acc[1][0], false, false);
    acc[1][1] = __builtin_amdgcn_wmma_f32_16x16x32_bf16(false, a1, false, b1, (short)0, acc[1][1], false, false);

    wait_async0();
    __syncthreads();
  }

  const int half = lane >> 4;
  const int ncol = lane & 15;
#pragma unroll
  for (int i = 0; i < 2; ++i) {
#pragma unroll
    for (int j = 0; j < 2; ++j) {
      int gn = (int)nwg + (2 * wb + j) * 16 + ncol;
      float bv = (bias1 ? bias1[gn] : 0.f) + (bias2 ? bias2[gn] : 0.f);
#pragma unroll
      for (int r = 0; r < 8; ++r) {
        size_t gm = mwg + (2 * wa + i) * 16 + r + 8 * half;
        C[gm * ldc + gn] = acc[i][j][r] + bv;
      }
    }
  }
}

// Persistent sequential scan: h_t = tanh(pre_t + h_{t-1} @ Whh^T).
// 8 workgroups; WG g owns output columns [g*128, g*128+128). Its 1024x128 bf16
// W_hh slice (256KB) is loaded once by the Tensor Data Mover and stays LDS-
// resident for all 512 steps. h ping-pongs through global (L2-resident).
#define NG_SCAN 8
#define SCAN_NC 128
__global__ __launch_bounds__(256) void rnn_scan_kernel(
    const float* __restrict__ pre,          // [B, S, H] f32
    const unsigned short* __restrict__ Whh, // [H, H] bf16
    unsigned short* __restrict__ hs,        // [B, S, H] bf16 out
    unsigned short* __restrict__ h_ping,    // [B, H] bf16
    unsigned short* __restrict__ h_pong,    // [B, H] bf16
    unsigned int* __restrict__ barrier_cnt) {
  extern __shared__ unsigned short Wls[];   // [SCAN_NC][H_] = 256KB

  const int g     = blockIdx.x;
  const int nbase = g * SCAN_NC;
  const int tid   = threadIdx.x;
  const int wave  = tid >> 5;
  const int lane  = tid & 31;

  // One wave issues a single TDM descriptor: 2D tile, rows nbase..nbase+127,
  // 1024 bf16 each, contiguous into LDS. Tracked by TENSORcnt.
  if (tid < 32) {
    unsigned long long ga = (unsigned long long)(uintptr_t)(Whh + (size_t)nbase * H_);
    unsigned int lds = (unsigned int)(uintptr_t)(&Wls[0]);
    u32x4 g0;
    g0[0] = 1u;                                            // count=1 valid user D#
    g0[1] = lds;                                           // lds_addr
    g0[2] = (unsigned int)(ga & 0xFFFFFFFFu);              // global_addr[31:0]
    g0[3] = (unsigned int)((ga >> 32) & 0x01FFFFFFu)       // global_addr[56:32]
            | (2u << 30);                                  // type = 2 (image)
    i32x8 g1;
    g1[0] = (int)(1u << 16);                               // data_size=1 (2B), mask=0
    g1[1] = (int)((H_ & 0xFFFFu) << 16);                   // tensor_dim0[15:0]
    g1[2] = (int)(((H_ >> 16) & 0xFFFFu) | (SCAN_NC << 16)); // dim0 hi | tensor_dim1 lo
    g1[3] = (int)(((SCAN_NC >> 16) & 0xFFFFu) | (H_ << 16)); // dim1 hi | tile_dim0=1024
    g1[4] = (int)(SCAN_NC & 0xFFFFu);                      // tile_dim1=128, tile_dim2=0
    g1[5] = (int)H_;                                       // tensor_dim0_stride lo
    g1[6] = 0;                                             // stride0 hi | stride1 lo
    g1[7] = 0;
    asm volatile("tensor_load_to_lds %0, %1" :: "s"(g0), "s"(g1) : "memory");
    __builtin_amdgcn_s_wait_tensorcnt(0);
  }
  // h0 = 0 for our column slice.
  for (int i = tid; i < 64 * SCAN_NC; i += 256) {
    int m = i >> 7, j = i & (SCAN_NC - 1);
    h_ping[(size_t)m * H_ + nbase + j] = 0;
  }
  __threadfence();
  unsigned int epoch = 0;
  grid_sync(barrier_cnt, epoch += NG_SCAN);

  const int wa   = wave >> 2;   // 0..1 -> m-tile pair (batch rows)
  const int wb   = wave & 3;    // 0..3 -> n-tile pair
  const int half = lane >> 4;
  const int ncol = lane & 15;

  for (int t = 0; t < S_; ++t) {
    const unsigned short* hp = (t & 1) ? h_pong : h_ping;
    unsigned short*       hn = (t & 1) ? h_ping : h_pong;

    v8f acc[2][2] = {};
    for (int kb = 0; kb < H_; kb += 32) {
      v16bf a0 = load_frag_a(hp, H_, (2 * wa + 0) * 16, kb, lane);
      v16bf a1 = load_frag_a(hp, H_, (2 * wa + 1) * 16, kb, lane);
      v16bf b0 = load_frag_b(Wls, H_, (2 * wb + 0) * 16, kb, lane);
      v16bf b1 = load_frag_b(Wls, H_, (2 * wb + 1) * 16, kb, lane);
      acc[0][0] = __builtin_amdgcn_wmma_f32_16x16x32_bf16(false, a0, false, b0, (short)0, acc[0][0], false, false);
      acc[0][1] = __builtin_amdgcn_wmma_f32_16x16x32_bf16(false, a0, false, b1, (short)0, acc[0][1], false, false);
      acc[1][0] = __builtin_amdgcn_wmma_f32_16x16x32_bf16(false, a1, false, b0, (short)0, acc[1][0], false, false);
      acc[1][1] = __builtin_amdgcn_wmma_f32_16x16x32_bf16(false, a1, false, b1, (short)0, acc[1][1], false, false);
    }

#pragma unroll
    for (int i = 0; i < 2; ++i) {
#pragma unroll
      for (int r = 0; r < 8; ++r) {
        int m = (2 * wa + i) * 16 + r + 8 * half;       // batch index
        size_t prerow = ((size_t)m * S_ + t) * H_;
#pragma unroll
        for (int j = 0; j < 2; ++j) {
          int gh = nbase + (2 * wb + j) * 16 + ncol;
          float v = tanhf(acc[i][j][r] + pre[prerow + gh]);
          unsigned short bv = f32_to_bf16(v);
          hs[prerow + gh] = bv;
          hn[(size_t)m * H_ + gh] = bv;
        }
      }
    }
    __threadfence();
    grid_sync(barrier_cnt, epoch += NG_SCAN);
  }
}

// ---------------- host side ----------------

extern "C" void kernel_launch(void* const* d_in, const int* in_sizes, int n_in,
                              void* d_out, int out_size, void* d_ws, size_t ws_size,
                              hipStream_t stream) {
  (void)in_sizes; (void)n_in; (void)out_size; (void)ws_size;

  const float* x    = (const float*)d_in[0];
  const float* Wih0 = (const float*)d_in[1];
  const float* Whh0 = (const float*)d_in[2];
  const float* bih0 = (const float*)d_in[3];
  const float* bhh0 = (const float*)d_in[4];
  const float* Wih1 = (const float*)d_in[5];
  const float* Whh1 = (const float*)d_in[6];
  const float* bih1 = (const float*)d_in[7];
  const float* bhh1 = (const float*)d_in[8];
  const float* fcw  = (const float*)d_in[9];
  const float* fcb  = (const float*)d_in[10];

  char* ws = (char*)d_ws;
  size_t off = 0;
  auto alloc = [&](size_t bytes) -> void* {
    void* p = ws + off;
    off += (bytes + 255) & ~(size_t)255;
    return p;
  };

  float*          pre   = (float*)alloc((size_t)B_ * S_ * H_ * 4);          // 128MB (reused)
  unsigned short* xbf   = (unsigned short*)alloc((size_t)B_ * S_ * I_ * 2); // 16MB
  unsigned short* hsbf  = (unsigned short*)alloc((size_t)B_ * S_ * H_ * 2); // 64MB (hs0 then hs1)
  unsigned short* wih0b = (unsigned short*)alloc((size_t)H_ * I_ * 2);
  unsigned short* whh0b = (unsigned short*)alloc((size_t)H_ * H_ * 2);
  unsigned short* wih1b = (unsigned short*)alloc((size_t)H_ * H_ * 2);
  unsigned short* whh1b = (unsigned short*)alloc((size_t)H_ * H_ * 2);
  unsigned short* fcwb  = (unsigned short*)alloc((size_t)O_ * H_ * 2);
  unsigned short* hping = (unsigned short*)alloc((size_t)B_ * H_ * 2);
  unsigned short* hpong = (unsigned short*)alloc((size_t)B_ * H_ * 2);
  unsigned int*   cnt   = (unsigned int*)alloc(256);

  auto cvt = [&](const float* s, unsigned short* d, size_t n) {
    unsigned nb = (unsigned)((n + 255) / 256);
    if (nb > 4096) nb = 4096;
    cvt_f32_bf16_kernel<<<dim3(nb), dim3(256), 0, stream>>>(s, d, n);
  };

  cvt(x,    xbf,   (size_t)B_ * S_ * I_);
  cvt(Wih0, wih0b, (size_t)H_ * I_);
  cvt(Whh0, whh0b, (size_t)H_ * H_);
  cvt(Wih1, wih1b, (size_t)H_ * H_);
  cvt(Whh1, whh1b, (size_t)H_ * H_);
  cvt(fcw,  fcwb,  (size_t)O_ * H_);

  dim3 blk(256);
  const int MT = (B_ * S_) / 128;           // 256 M-tiles of 128
  const size_t scan_lds = (size_t)SCAN_NC * H_ * 2;  // 256KB

  // Layer 0: pre0 = x @ Wih0^T + b_ih0 + b_hh0
  gemm_bf16_kernel<<<dim3(H_ / 64, MT), blk, 0, stream>>>(
      xbf, I_, wih0b, I_, pre, H_, I_, bih0, bhh0);
  hipMemsetAsync(cnt, 0, 256, stream);
  rnn_scan_kernel<<<dim3(NG_SCAN), blk, scan_lds, stream>>>(
      pre, whh0b, hsbf, hping, hpong, cnt);

  // Layer 1: pre1 = hs0 @ Wih1^T + b_ih1 + b_hh1
  gemm_bf16_kernel<<<dim3(H_ / 64, MT), blk, 0, stream>>>(
      hsbf, H_, wih1b, H_, pre, H_, H_, bih1, bhh1);
  hipMemsetAsync(cnt, 0, 256, stream);
  rnn_scan_kernel<<<dim3(NG_SCAN), blk, scan_lds, stream>>>(
      pre, whh1b, hsbf, hping, hpong, cnt);

  // Output: out = hs1 @ fc_w^T + fc_b
  gemm_bf16_kernel<<<dim3(O_ / 64, MT), blk, 0, stream>>>(
      hsbf, H_, fcwb, H_, (float*)d_out, O_, H_, fcb, nullptr);
}
